// CrossAttention_11751030522327
// MI455X (gfx1250) — matrix operand order repaired and checked by reading, not compiled, not run on hardware
//
#include <hip/hip_runtime.h>
#include <hip/hip_bf16.h>

// ---------------------------------------------------------------------------
// CDNA5 (gfx1250) transformer encoder layer, bf16 WMMA everywhere.
// GEMMs: LDS-tiled, 128x64 macro-tile per workgroup, 16x64 strip per wave
// (one A fragment feeds 4 v_wmma_f32_16x16x32_bf16). Weight k-slices are
// DOUBLE-BUFFERED through LDS with inline-asm GLOBAL_LOAD_ASYNC_TO_LDS_B128:
// step i+1 is prefetched while step i computes; s_wait_asynccnt 1 retires the
// older copy (async loads complete in order, CDNA5 ISA).
// Attention: per (b, 16-query rows) workgroup, loop heads: S = qp kp^T via
// WMMA -> LDS, row softmax, w_avg += P/H, ctx_h = P vp_h via WMMA.
// ---------------------------------------------------------------------------

typedef __attribute__((ext_vector_type(16))) __bf16 v16bf;
typedef __attribute__((ext_vector_type(8)))  __bf16 v8bf;
typedef __attribute__((ext_vector_type(8)))  float  v8f;

#define B_   8
#define LQ_  1024
#define LK_  1024
#define D_   512
#define H_   8
#define HD_  64
#define DFF_ 1024
#define NTOK (B_ * LQ_)
#define QT   16   // query rows per attention workgroup (16*1024*4B = 64KB LDS)

__device__ __forceinline__ void wait_async_le0()
{
#if defined(__has_builtin) && __has_builtin(__builtin_amdgcn_s_wait_asynccnt)
    __builtin_amdgcn_s_wait_asynccnt(0);
#else
    asm volatile("s_wait_asynccnt 0x0" ::: "memory");
#endif
}
__device__ __forceinline__ void wait_async_le1()
{
#if defined(__has_builtin) && __has_builtin(__builtin_amdgcn_s_wait_asynccnt)
    __builtin_amdgcn_s_wait_asynccnt(1);
#else
    asm volatile("s_wait_asynccnt 0x1" ::: "memory");
#endif
}

// Async copy of 16 bytes global -> LDS (per lane), tracked by ASYNCcnt.
// GV mode: VDST = LDS byte offset VGPR, VADDR = 64-bit global address pair.
__device__ __forceinline__ void async_copy_b128_to_lds(void* lds_generic,
                                                       const void* gsrc)
{
    unsigned lds_off =
        (unsigned)(size_t)(__attribute__((address_space(3))) void*)lds_generic;
    unsigned long long gaddr = (unsigned long long)(size_t)gsrc;
    asm volatile("global_load_async_to_lds_b128 %0, %1, off"
                 :: "v"(lds_off), "v"(gaddr)
                 : "memory");
}

// ---------------------------------------------------------------------------
// GEMM:  out[n,m] = act( (sum_k A[n,k] * W[m,k]) + bias[m] ) * scale
// A: [Nrows,K] bf16 row-major, W: [M,K] bf16 row-major (computes A W^T).
// Block (256 thr / 8 waves): 128 rows x 64 cols. Wave w: rows [n0+16w, +16),
// all 64 cols -> 4 C tiles sharing one A fragment per 32-wide k-step.
// W k-slice (64 cols x 32 k = 4KB) ping-pong staged in LDS via async DMA.
// ---------------------------------------------------------------------------
__global__ __launch_bounds__(256)
void gemm_wmma_bf16(const __bf16* __restrict__ A, const __bf16* __restrict__ W,
                    const float* __restrict__ bias,
                    float* __restrict__ outF, __bf16* __restrict__ outB,
                    int Nrows, int K, int M, float scaleOut, int doRelu)
{
    __shared__ __attribute__((aligned(32))) __bf16 Wl[2][64 * 32]; // 2x4KB ping-pong

    const int lane = threadIdx.x & 31;
    const int wave = threadIdx.x >> 5;
    const int blocksM = M >> 6;
    const int m0 = (blockIdx.x % blocksM) << 6;
    const int n0 = ((blockIdx.x / blocksM) << 7) + (wave << 4);

    const int row    = lane & 15;    // A row within tile / B column within tile
    const int hiHalf = lane >> 4;

    // A fragment addressing (16-bit A 16x32, contiguous K to match B):
    // lanes 0-15 hold K {0..15}; lanes 16-31 hold {16..31}
    const __bf16* aRow = A + (size_t)(n0 + row) * K + (hiHalf ? 16 : 0);

    // LDS staging map: thread t copies 8 bf16 (16B) of the 64x32 W slice
    const int tcol = threadIdx.x >> 2;            // 0..63 -> W row m0+tcol
    const int tk   = (threadIdx.x & 3) << 3;      // 0,8,16,24
    const __bf16* wsrc  = W + (size_t)(m0 + tcol) * K + tk;
    __bf16*       wdst0 = &Wl[0][tcol * 32 + tk];
    __bf16*       wdst1 = &Wl[1][tcol * 32 + tk];

    v8f acc[4] = {v8f{}, v8f{}, v8f{}, v8f{}};

    const int steps = K >> 5;
    async_copy_b128_to_lds(wdst0, wsrc);               // prologue: stage step 0

    for (int i = 0; i < steps; ++i) {
        const int k0 = i << 5;

        // ---- prefetch step i+1 into the alternate buffer, retire step i ----
        if (i + 1 < steps) {
            async_copy_b128_to_lds((i & 1) ? wdst0 : wdst1, wsrc + k0 + 32);
            wait_async_le1();   // in-order completion => buffer i is resident
        } else {
            wait_async_le0();
        }
        __syncthreads();        // all waves' copies of buffer i visible

        // ---- A fragment (direct from global) ----
        v8bf a0 = *(const v8bf*)(aRow + k0);
        v8bf a1 = *(const v8bf*)(aRow + k0 + 8);
        v16bf a;
#pragma unroll
        for (int ii = 0; ii < 8; ++ii) { a[ii] = a0[ii]; a[8 + ii] = a1[ii]; }

        // ---- 4 WMMAs reuse the A fragment across the 64-col strip ----
        const __bf16* Wbuf = Wl[i & 1];
#pragma unroll
        for (int t = 0; t < 4; ++t) {
            // B fragment: lane=col, 16 contiguous K (lanes 0-15: K0..15, 16-31: K16..31)
            v16bf b = *(const v16bf*)&Wbuf[(t * 16 + row) * 32 + (hiHalf ? 16 : 0)];
            acc[t] = __builtin_amdgcn_wmma_f32_16x16x32_bf16(
                         false, a, false, b, (short)0, acc[t], false, false);
        }
        __syncthreads();        // reads of buffer i done before it is overwritten
    }

    const int rowAdd = hiHalf ? 8 : 0;
#pragma unroll
    for (int t = 0; t < 4; ++t) {
        const int  mcol = m0 + t * 16 + row;
        const float bb  = bias ? bias[mcol] : 0.0f;
#pragma unroll
        for (int r = 0; r < 8; ++r) {
            float v = (acc[t][r] + bb) * scaleOut;
            if (doRelu) v = fmaxf(v, 0.0f);
            size_t idx = (size_t)(n0 + r + rowAdd) * M + mcol;
            if (outF) outF[idx] = v;
            if (outB) outB[idx] = (__bf16)v;
        }
    }
}

// ---------------------------------------------------------------------------
// Fused attention for one (batch, 16-query tile), looping over heads.
// LDS holds the full 16 x LK score/probability tile (64 KB).
// ---------------------------------------------------------------------------
__global__ __launch_bounds__(256)
void attention_wmma(const __bf16* __restrict__ qp,   // [B,LQ,D] pre-scaled
                    const __bf16* __restrict__ kp,   // [B,LK,D]
                    const __bf16* __restrict__ vpT,  // [B,H,HD,LK]
                    __bf16* __restrict__ ctx,        // [B,LQ,D]
                    float* __restrict__ wavg)        // [B,LQ,LK]
{
    __shared__ float S[QT][LK_];           // 64 KB of the WGP's 320 KB LDS

    const int b      = blockIdx.y;
    const int q0     = blockIdx.x * QT;
    const int lane   = threadIdx.x & 31;
    const int wave   = threadIdx.x >> 5;
    const int row    = lane & 15;
    const int hiHalf = lane >> 4;
    const int rowAdd = hiHalf ? 8 : 0;

    for (int h = 0; h < H_; ++h) {
        // ---- phase 1: S = qp_tile @ kp_head^T  (64 col-tiles over 8 waves) ----
        const __bf16* aRow = qp + (size_t)(b * LQ_ + q0 + row) * D_ + h * HD_
                                + (hiHalf ? 16 : 0);
        for (int ct = wave; ct < (LK_ >> 4); ct += 8) {
            const int n0 = ct << 4;
            const __bf16* bRow = kp + (size_t)(b * LK_ + n0 + row) * D_ + h * HD_
                                    + (hiHalf ? 16 : 0);
            v8f acc = {};
#pragma unroll
            for (int k0 = 0; k0 < HD_; k0 += 32) {
                v8bf a0 = *(const v8bf*)(aRow + k0);
                v8bf a1 = *(const v8bf*)(aRow + k0 + 8);
                v16bf a;
#pragma unroll
                for (int i = 0; i < 8; ++i) { a[i] = a0[i]; a[8 + i] = a1[i]; }
                v16bf bf = *(const v16bf*)(bRow + k0);
                acc = __builtin_amdgcn_wmma_f32_16x16x32_bf16(
                          false, a, false, bf, (short)0, acc, false, false);
            }
#pragma unroll
            for (int r = 0; r < 8; ++r)
                S[r + rowAdd][n0 + row] = acc[r];
        }
        __syncthreads();

        // ---- phase 2: row softmax (one wave per row, wave32 shuffles) ----
        for (int rr = wave; rr < QT; rr += 8) {
            float m = -1e30f;
            for (int j = lane; j < LK_; j += 32) m = fmaxf(m, S[rr][j]);
#pragma unroll
            for (int o = 16; o >= 1; o >>= 1) m = fmaxf(m, __shfl_xor(m, o, 32));
            float s = 0.0f;
            for (int j = lane; j < LK_; j += 32) {
                float e = __expf(S[rr][j] - m);
                S[rr][j] = e;
                s += e;
            }
#pragma unroll
            for (int o = 16; o >= 1; o >>= 1) s += __shfl_xor(s, o, 32);
            const float inv = 1.0f / s;
            for (int j = lane; j < LK_; j += 32) S[rr][j] *= inv;
        }
        __syncthreads();

        // ---- w_avg accumulation (block owns its [q0,q0+QT) x LK slab) ----
        for (int i = threadIdx.x; i < QT * LK_; i += 256) {
            const int rr = i >> 10;          // LK_ == 1024
            const int j  = i & (LK_ - 1);
            const size_t g = ((size_t)b * LQ_ + q0 + rr) * LK_ + j;
            const float  p = S[rr][j] * (1.0f / H_);
            wavg[g] = (h == 0) ? p : (wavg[g] + p);
        }

        // ---- phase 3: ctx_h = P(16 x LK) @ vp_h(LK x HD); waves 0..3 ----
        if (wave < 4) {
            const int c0 = wave << 4;
            const __bf16* bRow = vpT + ((size_t)(b * H_ + h) * HD_ + c0 + row) * LK_
                                     + (hiHalf ? 16 : 0);
            v8f acc = {};
            for (int k0 = 0; k0 < LK_; k0 += 32) {
                const float* pr = &S[row][k0 + (hiHalf ? 16 : 0)];
                v16bf a;
#pragma unroll
                for (int i = 0; i < 8; ++i) {
                    a[i]     = (__bf16)pr[i];
                    a[8 + i] = (__bf16)pr[i + 8];
                }
                v16bf bf = *(const v16bf*)(bRow + k0);
                acc = __builtin_amdgcn_wmma_f32_16x16x32_bf16(
                          false, a, false, bf, (short)0, acc, false, false);
            }
#pragma unroll
            for (int r = 0; r < 8; ++r)
                ctx[(size_t)(b * LQ_ + q0 + r + rowAdd) * D_ + h * HD_ + c0 + row] =
                    (__bf16)acc[r];
        }
        __syncthreads();   // LDS reused by next head
    }
}

// ---------------------------------------------------------------------------
// y = LN(a + b) * g + beta ; optional fp32 and bf16 outputs. Wave per row.
// ---------------------------------------------------------------------------
__global__ __launch_bounds__(256)
void add_layernorm(const float* __restrict__ A, const float* __restrict__ Badd,
                   const float* __restrict__ g, const float* __restrict__ beta,
                   float* __restrict__ outF, __bf16* __restrict__ outB, int rows)
{
    const int r = blockIdx.x * 8 + (threadIdx.x >> 5);
    if (r >= rows) return;
    const int lane = threadIdx.x & 31;
    const float* pa = A    + (size_t)r * D_;
    const float* pb = Badd + (size_t)r * D_;

    float s1 = 0.0f, s2 = 0.0f;
    for (int j = lane; j < D_; j += 32) {
        const float t = pa[j] + pb[j];
        s1 += t; s2 += t * t;
    }
#pragma unroll
    for (int o = 16; o >= 1; o >>= 1) {
        s1 += __shfl_xor(s1, o, 32);
        s2 += __shfl_xor(s2, o, 32);
    }
    const float mean = s1 * (1.0f / D_);
    const float var  = s2 * (1.0f / D_) - mean * mean;
    const float rstd = rsqrtf(var + 1e-5f);
    for (int j = lane; j < D_; j += 32) {
        const float t = pa[j] + pb[j];
        const float y = (t - mean) * rstd * g[j] + beta[j];
        if (outF) outF[(size_t)r * D_ + j] = y;
        if (outB) outB[(size_t)r * D_ + j] = (__bf16)y;
    }
}

// ---------------------------------------------------------------------------
__global__ __launch_bounds__(256)
void f2bf(const float* __restrict__ in, __bf16* __restrict__ out, int n)
{
    for (int i = blockIdx.x * 256 + threadIdx.x; i < n; i += gridDim.x * 256)
        out[i] = (__bf16)in[i];
}

// vp [B,LK,D] -> vpT [B,H,HD,LK] so WMMA B-fragments are contiguous in K.
__global__ __launch_bounds__(256)
void transpose_vp(const __bf16* __restrict__ vp, __bf16* __restrict__ vpT)
{
    const int total = B_ * LK_ * D_;
    for (int i = blockIdx.x * 256 + threadIdx.x; i < total; i += gridDim.x * 256) {
        const int d  = i % D_;
        const int kk = (i / D_) % LK_;
        const int b  = i / (D_ * LK_);
        const int h  = d / HD_;
        const int dd = d % HD_;
        vpT[((size_t)(b * H_ + h) * HD_ + dd) * LK_ + kk] = vp[i];
    }
}

// ---------------------------------------------------------------------------
extern "C" void kernel_launch(void* const* d_in, const int* in_sizes, int n_in,
                              void* d_out, int out_size, void* d_ws, size_t ws_size,
                              hipStream_t stream)
{
    (void)in_sizes; (void)n_in; (void)out_size; (void)ws_size;

    const float* q         = (const float*)d_in[0];
    const float* k         = (const float*)d_in[1];
    const float* v         = (const float*)d_in[2];
    const float* in_proj_w = (const float*)d_in[3];
    const float* in_proj_b = (const float*)d_in[4];
    const float* out_w     = (const float*)d_in[5];
    const float* out_b     = (const float*)d_in[6];
    const float* ln1_g     = (const float*)d_in[7];
    const float* ln1_b     = (const float*)d_in[8];
    const float* w1        = (const float*)d_in[9];
    const float* b1        = (const float*)d_in[10];
    const float* w2        = (const float*)d_in[11];
    const float* b2        = (const float*)d_in[12];
    const float* ln2_g     = (const float*)d_in[13];
    const float* ln2_b     = (const float*)d_in[14];

    float* x_out = (float*)d_out;                       // [NTOK, D]
    float* wavg  = x_out + (size_t)NTOK * D_;           // [B, LQ, LK]

    // ---- workspace carve-up (256B aligned) ----
    char* wp = (char*)d_ws;
    auto alloc_bf = [&](size_t n) -> __bf16* {
        __bf16* p = (__bf16*)wp; wp += ((n * 2 + 255) / 256) * 256; return p; };
    auto alloc_f = [&](size_t n) -> float* {
        float* p = (float*)wp; wp += ((n * 4 + 255) / 256) * 256; return p; };

    const size_t ND = (size_t)NTOK * D_;
    __bf16* q_bf   = alloc_bf(ND);
    __bf16* k_bf   = alloc_bf(ND);
    __bf16* v_bf   = alloc_bf(ND);
    __bf16* ipw_bf = alloc_bf((size_t)3 * D_ * D_);     // [Wq;Wk;Wv]
    __bf16* wo_bf  = alloc_bf((size_t)D_ * D_);
    __bf16* w1_bf  = alloc_bf((size_t)DFF_ * D_);
    __bf16* w2_bf  = alloc_bf((size_t)D_ * DFF_);
    __bf16* qp_bf  = alloc_bf(ND);
    __bf16* kp_bf  = alloc_bf(ND);
    __bf16* vp_bf  = alloc_bf(ND);
    __bf16* vpT_bf = alloc_bf(ND);
    __bf16* ctx_bf = alloc_bf(ND);
    __bf16* x1_bf  = alloc_bf(ND);
    __bf16* ffh_bf = alloc_bf((size_t)NTOK * DFF_);
    float*  attn_f = alloc_f(ND);
    float*  x1_f   = alloc_f(ND);
    float*  ff2_f  = alloc_f(ND);

    const dim3 blk(256);

    // ---- fp32 -> bf16 converts ----
    f2bf<<<2048, blk, 0, stream>>>(q, q_bf, (int)ND);
    f2bf<<<2048, blk, 0, stream>>>(k, k_bf, (int)ND);
    f2bf<<<2048, blk, 0, stream>>>(v, v_bf, (int)ND);
    f2bf<<<1024, blk, 0, stream>>>(in_proj_w, ipw_bf, 3 * D_ * D_);
    f2bf<<<512,  blk, 0, stream>>>(out_w, wo_bf, D_ * D_);
    f2bf<<<512,  blk, 0, stream>>>(w1, w1_bf, DFF_ * D_);
    f2bf<<<512,  blk, 0, stream>>>(w2, w2_bf, D_ * DFF_);

    // grid = (Nrows/128) * (M/64) macro-tiles
    auto gblocks = [](int Nrows, int M) { return (Nrows / 128) * (M / 64); };

    // ---- QKV projections (scale 1/sqrt(hd)=0.125 folded into Q) ----
    gemm_wmma_bf16<<<gblocks(NTOK, D_), blk, 0, stream>>>(
        q_bf, ipw_bf,                 in_proj_b,          nullptr, qp_bf,
        NTOK, D_, D_, 0.125f, 0);
    gemm_wmma_bf16<<<gblocks(NTOK, D_), blk, 0, stream>>>(
        k_bf, ipw_bf + (size_t)D_ * D_,     in_proj_b + D_,     nullptr, kp_bf,
        NTOK, D_, D_, 1.0f, 0);
    gemm_wmma_bf16<<<gblocks(NTOK, D_), blk, 0, stream>>>(
        v_bf, ipw_bf + (size_t)2 * D_ * D_, in_proj_b + 2 * D_, nullptr, vp_bf,
        NTOK, D_, D_, 1.0f, 0);

    transpose_vp<<<2048, blk, 0, stream>>>(vp_bf, vpT_bf);

    // ---- fused attention + w_avg ----
    dim3 agrid(LQ_ / QT, B_);
    attention_wmma<<<agrid, blk, 0, stream>>>(qp_bf, kp_bf, vpT_bf, ctx_bf, wavg);

    // ---- output projection, LN1 ----
    gemm_wmma_bf16<<<gblocks(NTOK, D_), blk, 0, stream>>>(
        ctx_bf, wo_bf, out_b, attn_f, nullptr, NTOK, D_, D_, 1.0f, 0);
    add_layernorm<<<NTOK / 8, blk, 0, stream>>>(
        q, attn_f, ln1_g, ln1_b, x1_f, x1_bf, NTOK);

    // ---- FFN: relu(x1 W1^T + b1) W2^T + b2, then LN2 into d_out ----
    gemm_wmma_bf16<<<gblocks(NTOK, DFF_), blk, 0, stream>>>(
        x1_bf, w1_bf, b1, nullptr, ffh_bf, NTOK, D_, DFF_, 1.0f, 1);
    gemm_wmma_bf16<<<gblocks(NTOK, D_), blk, 0, stream>>>(
        ffh_bf, w2_bf, b2, ff2_f, nullptr, NTOK, DFF_, D_, 1.0f, 0);
    add_layernorm<<<NTOK / 8, blk, 0, stream>>>(
        x1_f, ff2_f, ln2_g, ln2_b, x_out, nullptr, NTOK);
}
